// PFNN_41008347742352
// MI455X (gfx1250) — matrix operand order repaired and checked by reading, not compiled, not run on hardware
//
#include <hip/hip_runtime.h>

typedef _Float16 half_t;
typedef __attribute__((ext_vector_type(16))) _Float16 v16h;
typedef __attribute__((ext_vector_type(8)))  float    v8f;

#define NPTS   1048576
#define NTILES (NPTS / 16)

// Compiler-only fence: keeps the cross-lane LDS store->load pipeline intact
// (LLVM's per-thread AA cannot see lane exchange through LDS and otherwise
// hoists the fragment reloads / kills the producer stores).
#define LDS_PIPELINE_FENCE() asm volatile("" ::: "memory")

__device__ __forceinline__ float fast_tanh(float v) {
#if __has_builtin(__builtin_amdgcn_tanhf)
    return __builtin_amdgcn_tanhf(v);          // v_tanh_f32 (CDNA5 trans op)
#else
    float e = __expf(2.0f * v);
    return (e - 1.0f) / (e + 1.0f);
#endif
}

union F16Frag {
    v16h v;
    unsigned u[8];
    _Float16 h[16];
};

__launch_bounds__(256)
__global__ void pfnn_wmma_kernel(const float* __restrict__ x,
                                 const float* __restrict__ W0, const float* __restrict__ b0,
                                 const float* __restrict__ W1, const float* __restrict__ b1,
                                 const float* __restrict__ W2, const float* __restrict__ b2,
                                 const float* __restrict__ W3, const float* __restrict__ b3,
                                 float* __restrict__ out, int nblocks)
{
    // ---- LDS staging: weights (f16, transposed [b][n][k]) + small f32 tables ----
    __shared__ __align__(16) _Float16 sW1[4 * 32 * 32];   // [b][n][k]
    __shared__ __align__(16) _Float16 sW2[4 * 32 * 32];   // [b][n][k]
    __shared__ __align__(16) float    sW0[4 * 3 * 32];    // [b][d][k]
    __shared__ __align__(16) float    sB0[128], sB1[128], sB2[128], sW3[128];
    __shared__ __align__(16) float    sB3[4];
    // per-wave transpose scratch: 16 rows x stride 34 halves (68B rows, 4B aligned)
    __shared__ __align__(16) _Float16 sH[8][16 * 34];

    const int tid = threadIdx.x;
    for (int i = tid; i < 4096; i += 256) {
        int bb = i >> 10, k = (i >> 5) & 31, n = i & 31;
        sW1[(bb * 32 + n) * 32 + k] = (half_t)W1[i];
        sW2[(bb * 32 + n) * 32 + k] = (half_t)W2[i];
    }
    for (int i = tid; i < 384; i += 256) sW0[i] = W0[i];
    if (tid < 128) {
        sB0[tid] = b0[tid];
        sB1[tid] = b1[tid];
        sB2[tid] = b2[tid];
        sW3[tid] = W3[tid];                 // W3 is [b][k][1] -> flat [b*32+k]
    }
    if (tid < 4) sB3[tid] = b3[tid];
    __syncthreads();

    const int lane = tid & 31;
    const int wave = tid >> 5;
    const int half = lane >> 4;             // K-half of A fragment this lane holds
    const int lrow = lane & 15;             // A: row m; B/C/D: column n
    half_t* hb = &sH[wave][0];

    // A-layout K index per slot s = j*2+e (16-bit A 16x32 table, CDNA5 ISA 7.12.2)
    int kA[16];
#pragma unroll
    for (int j = 0; j < 8; ++j)
#pragma unroll
        for (int e = 0; e < 2; ++e)
            kA[j * 2 + e] = ((j >> 2) << 4) + (half << 3) + ((j & 3) << 1) + e;

    // wave id is wave-uniform in reality; make it uniform for the compiler too,
    // so the tile loop is a scalar (s_cbranch) loop and EXEC stays all-ones
    // around the WMMAs.
    const int wglobal = __builtin_amdgcn_readfirstlane(blockIdx.x * 8 + wave);
    const int wstride = nblocks * 8;

    for (int b = 0; b < 4; ++b) {
        // ---- per-branch register setup (hoisted out of the tile loop) ----
        float w0v[16][3], b0v[16], w3v[16];
#pragma unroll
        for (int s = 0; s < 16; ++s) {
            int k = kA[s];
            w0v[s][0] = sW0[b * 96 +      k];
            w0v[s][1] = sW0[b * 96 + 32 + k];
            w0v[s][2] = sW0[b * 96 + 64 + k];
            b0v[s]    = sB0[b * 32 + k];
            w3v[s]    = sW3[b * 32 + k];
        }
        float bias1[2], bias2[2];
        bias1[0] = sB1[b * 32 + lrow];  bias1[1] = sB1[b * 32 + 16 + lrow];
        bias2[0] = sB2[b * 32 + lrow];  bias2[1] = sB2[b * 32 + 16 + lrow];
        const float b3v = sB3[b];

        // Weight B-fragments (B 32x16 16-bit layout: lane=n, vgpr j holds k = kA[2j],+1)
        F16Frag w1f[2], w2f[2];
#pragma unroll
        for (int nt = 0; nt < 2; ++nt) {
            int n = nt * 16 + lrow;
#pragma unroll
            for (int j = 0; j < 8; ++j) {
                w1f[nt].u[j] = *(const unsigned*)&sW1[(b * 32 + n) * 32 + kA[j * 2]];
                w2f[nt].u[j] = *(const unsigned*)&sW2[(b * 32 + n) * 32 + kA[j * 2]];
            }
        }

        for (int t = wglobal; t < NTILES; t += wstride) {
            const int base = t * 16;
            const float* xp = x + (size_t)(base + lrow) * 3;
            const float x0 = xp[0], x1 = xp[1], x2 = xp[2];

            // ---- layer 0 (K=3): fp32 VALU straight into WMMA A layout ----
            v16h a1;
#pragma unroll
            for (int s = 0; s < 16; ++s) {
                float h = fast_tanh(fmaf(x0, w0v[s][0],
                                   fmaf(x1, w0v[s][1],
                                   fmaf(x2, w0v[s][2], b0v[s]))));
                a1[s] = (half_t)h;
            }

            // ---- layer 1: two 16x16x32 f16 WMMAs ----
            v8f c0 = {}, c1 = {};
            c0 = __builtin_amdgcn_wmma_f32_16x16x32_f16(false, a1, false, w1f[0].v, (short)0, c0, false, false);
            c1 = __builtin_amdgcn_wmma_f32_16x16x32_f16(false, a1, false, w1f[1].v, (short)0, c1, false, false);

            // bias + tanh, spill to per-wave LDS in [m][k] for the D->A transpose
            LDS_PIPELINE_FENCE();
#pragma unroll
            for (int r = 0; r < 8; ++r) {
                int m = half * 8 + r;
                hb[m * 34 + lrow]      = (half_t)fast_tanh(c0[r] + bias1[0]);
                hb[m * 34 + 16 + lrow] = (half_t)fast_tanh(c1[r] + bias1[1]);
            }
            LDS_PIPELINE_FENCE();
            F16Frag a2;
#pragma unroll
            for (int j = 0; j < 8; ++j)
                a2.u[j] = *(const unsigned*)&hb[lrow * 34 + kA[j * 2]];
            LDS_PIPELINE_FENCE();

            // ---- layer 2: two 16x16x32 f16 WMMAs ----
            v8f d0 = {}, d1 = {};
            d0 = __builtin_amdgcn_wmma_f32_16x16x32_f16(false, a2.v, false, w2f[0].v, (short)0, d0, false, false);
            d1 = __builtin_amdgcn_wmma_f32_16x16x32_f16(false, a2.v, false, w2f[1].v, (short)0, d1, false, false);

#pragma unroll
            for (int r = 0; r < 8; ++r) {
                int m = half * 8 + r;
                hb[m * 34 + lrow]      = (half_t)fast_tanh(d0[r] + bias2[0]);
                hb[m * 34 + 16 + lrow] = (half_t)fast_tanh(d1[r] + bias2[1]);
            }
            LDS_PIPELINE_FENCE();
            F16Frag a3;
#pragma unroll
            for (int j = 0; j < 8; ++j)
                a3.u[j] = *(const unsigned*)&hb[lrow * 34 + kA[j * 2]];
            LDS_PIPELINE_FENCE();

            // ---- layer 3 (32 -> 1): per-lane dot over A slots + half-combine ----
            float partial = 0.0f;
#pragma unroll
            for (int s = 0; s < 16; ++s)
                partial = fmaf((float)a3.h[s], w3v[s], partial);
            float total = partial + __shfl_xor(partial, 16, 32);

            if (half == 0)
                out[(size_t)(base + lrow) * 4 + b] = total + b3v;
        }
    }
}

extern "C" void kernel_launch(void* const* d_in, const int* in_sizes, int n_in,
                              void* d_out, int out_size, void* d_ws, size_t ws_size,
                              hipStream_t stream) {
    const float* x  = (const float*)d_in[0];
    const float* W0 = (const float*)d_in[1];
    const float* b0 = (const float*)d_in[2];
    const float* W1 = (const float*)d_in[3];
    const float* b1 = (const float*)d_in[4];
    const float* W2 = (const float*)d_in[5];
    const float* b2 = (const float*)d_in[6];
    const float* W3 = (const float*)d_in[7];
    const float* b3 = (const float*)d_in[8];
    float* outp = (float*)d_out;

    const int nblocks = 2048;   // 16384 waves -> 4 tiles per wave per branch
    pfnn_wmma_kernel<<<nblocks, 256, 0, stream>>>(x, W0, b0, W1, b1, W2, b2, W3, b3,
                                                  outp, nblocks);
    (void)in_sizes; (void)n_in; (void)out_size; (void)d_ws; (void)ws_size;
}
